// Bilstm_crf_9079560864163
// MI455X (gfx1250) — compile-verified, matrix operand order checked by reading, not run on hardware
//
#include <hip/hip_runtime.h>
#include <cstdint>
#include <cstddef>

// ---------------- problem constants ----------------
#define EMB   128
#define HID   256
#define G4    1024      // 4*HID (gates i,f,g,o)
#define NTAG  9
#define BATCH 64
#define SEQ   512
#define ROWS  (BATCH*SEQ)   // 32768

typedef __attribute__((ext_vector_type(16))) _Float16 v16h;
typedef __attribute__((ext_vector_type(8)))  _Float16 v8h;
typedef __attribute__((ext_vector_type(8)))  float    v8f;

__device__ __forceinline__ v8f wmma_f16(v16h a, v16h b, v8f c) {
  // D = A(16x32) * B(32x16) + C, f32 accumulate
  return __builtin_amdgcn_wmma_f32_16x16x32_f16(
      /*neg_a=*/false, a, /*neg_b=*/false, b,
      /*c_mod=*/(short)0, c, /*reuse_a=*/false, /*reuse_b=*/false);
}

// 16x32 fragment (rows x K), row-major source, element stride `stride`.
// Per ISA 16-bit layout: lane L -> row = L&15, half = L>>4;
// lane elements cover K = [k0+8*half, +8) and K = [k0+16+8*half, +8)
// => two contiguous 16-byte chunks per lane (compiler merges to b128 loads).
__device__ __forceinline__ v16h load_frag(const _Float16* base, int stride,
                                          int row0, int k0, int lane) {
  const int r   = row0 + (lane & 15);
  const int hlf = lane >> 4;
  const _Float16* p = base + (size_t)r * stride + k0 + 8 * hlf;
  v8h lo = *(const v8h*)p;
  v8h hi = *(const v8h*)(p + 16);
  v16h out;
#pragma unroll
  for (int i = 0; i < 8; ++i) { out[i] = lo[i]; out[i + 8] = hi[i]; }
  return out;
}

__device__ __forceinline__ float sigf(float x) { return 1.f / (1.f + __expf(-x)); }

// --------- CDNA5 async global->LDS copy (ASYNCcnt-tracked, bypasses VGPRs) ---------
// __shared__ objects are AS(3); the generic pointer's low 32 bits are the LDS
// byte offset (flat LDS aperture truncates to addr[31:0]).
__device__ __forceinline__ uint32_t lds_byte_addr(const void* p) {
  return (uint32_t)(uintptr_t)p;
}
__device__ __forceinline__ void async_copy_b128(uint32_t lds_addr, const void* gaddr) {
  asm volatile("global_load_async_to_lds_b128 %0, %1, off"
               :: "v"(lds_addr), "v"(gaddr)
               : "memory");
}
__device__ __forceinline__ void wait_asynccnt0() {
  asm volatile("s_wait_asynccnt 0x0" ::: "memory");
}

// ---------------- small prep kernels ----------------
__global__ void cast_f32_f16_kernel(const float* __restrict__ in,
                                    _Float16* __restrict__ out, int n) {
  int i = blockIdx.x * blockDim.x + threadIdx.x;
  if (i < n) out[i] = (_Float16)in[i];
}

// lin_w [9,512] -> padded f16 [16,512] (rows 9..15 zero)
__global__ void pad_linw_kernel(const float* __restrict__ lin_w,
                                _Float16* __restrict__ out) {
  int i = blockIdx.x * blockDim.x + threadIdx.x;
  if (i >= 16 * 512) return;
  int r = i >> 9, c = i & 511;
  out[i] = (r < NTAG) ? (_Float16)lin_w[r * 512 + c] : (_Float16)0.f;
}

__global__ void embed_kernel(const int* __restrict__ idx,
                             const float* __restrict__ emb,
                             _Float16* __restrict__ X) {
  int i = blockIdx.x * blockDim.x + threadIdx.x;   // over ROWS*EMB
  if (i >= ROWS * EMB) return;
  int row = i >> 7, col = i & 127;
  X[i] = (_Float16)emb[(size_t)idx[row] * EMB + col];
}

// ---------------- input projection: xp = X @ W_ih^T + (b_ih+b_hh) ----------------
// grid: (4096, 2) x 256 thr. blockIdx.y = direction. Each wave: 16x64 tile, K=128.
__global__ __launch_bounds__(256) void input_proj_kernel(
    const _Float16* __restrict__ X,
    const _Float16* __restrict__ wih_f, const _Float16* __restrict__ wih_b,
    const float* __restrict__ bih_f, const float* __restrict__ bhh_f,
    const float* __restrict__ bih_b, const float* __restrict__ bhh_b,
    _Float16* __restrict__ xp_f, _Float16* __restrict__ xp_b) {
  const int dir = blockIdx.y;
  const _Float16* wih = dir ? wih_b : wih_f;
  const float* bih = dir ? bih_b : bih_f;
  const float* bhh = dir ? bhh_b : bhh_f;
  _Float16* xp = dir ? xp_b : xp_f;

  const int lane = threadIdx.x & 31;
  const int wave = threadIdx.x >> 5;
  const int w  = blockIdx.x * 8 + wave;
  const int m0 = (w >> 4) * 16;        // 2048 M-tiles
  const int n0 = (w & 15) * 64;        // 16 N-supertiles of 64
  const int hlf = lane >> 4, ln = lane & 15;

  v8f acc[4] = {};
#pragma unroll
  for (int k = 0; k < 4; ++k) {
    v16h a = load_frag(X, EMB, m0, k * 32, lane);
#pragma unroll
    for (int n = 0; n < 4; ++n) {
      v16h b = load_frag(wih, EMB, n0 + n * 16, k * 32, lane);  // row n of W_ih = col n of W_ih^T
      acc[n] = wmma_f16(a, b, acc[n]);
    }
  }
#pragma unroll
  for (int n = 0; n < 4; ++n) {
    const int col = n0 + n * 16 + ln;
    const float bias = bih[col] + bhh[col];
#pragma unroll
    for (int r = 0; r < 8; ++r) {
      const int row = m0 + r + 8 * hlf;
      xp[(size_t)row * G4 + col] = (_Float16)(acc[n][r] + bias);
    }
  }
}

// ---------------- LSTM recurrence (persistent per dir x batch-tile) ----------------
// grid: 8 blocks x 256 thr. block = (dir, batch tile of 16 rows).
// Per step:
//   1) issue async DMA: xp[t] (32KB) -> gate LDS buffer   (ASYNCcnt)
//   2) concurrently: acc = h(LDS,f16) @ W_hh^T via WMMA (W_hh streamed from L2)
//   3) s_wait_asynccnt 0 + barrier; gates += acc (each element owned by one lane)
//   4) elementwise LSTM cell; h -> LDS + hcat(global)
__global__ __launch_bounds__(256) void lstm_rec_kernel(
    const _Float16* __restrict__ xp_f, const _Float16* __restrict__ xp_b,
    const _Float16* __restrict__ whh_f, const _Float16* __restrict__ whh_b,
    _Float16* __restrict__ hcat /* [ROWS,512]: fwd cols 0..255, bwd 256..511 */) {
  const int dir = blockIdx.x >> 2;
  const int b0  = (blockIdx.x & 3) * 16;
  const _Float16* xp  = dir ? xp_b : xp_f;
  const _Float16* whh = dir ? whh_b : whh_f;

  __shared__ _Float16 h_lds[16 * 264];    // [16][256] (+pad)
  __shared__ _Float16 g_lds[16 * 1032];   // [16][1024] gates (+pad)

  const int tid = threadIdx.x;
  const int lane = tid & 31, wave = tid >> 5;
  const int hlf = lane >> 4, ln = lane & 15;
  const int cr = tid & 15;      // cell phase: batch row
  const int cg = tid >> 4;      // cell phase: 16-col group

  for (int i = tid; i < 16 * 264; i += 256) h_lds[i] = (_Float16)0.f;
  float cst[16];
#pragma unroll
  for (int j = 0; j < 16; ++j) cst[j] = 0.f;

  // async-copy addressing (fixed per thread): row = tid>>4, 64-col chunk = tid&15
  const int arow = tid >> 4, achunk = tid & 15;
  const uint32_t gdst = lds_byte_addr(g_lds + arow * 1032 + achunk * 64);
  __syncthreads();

  for (int step = 0; step < SEQ; ++step) {
    const int t = dir ? (SEQ - 1 - step) : step;

    // 1) launch async xp[t] -> g_lds (8 x b128 per thread); no barrier needed:
    //    gates are not read until after the WMMA block below.
    {
      const _Float16* src = xp + ((size_t)(b0 + arow) * SEQ + t) * G4 + achunk * 64;
#pragma unroll
      for (int v = 0; v < 8; ++v)
        async_copy_b128(gdst + v * 16, src + v * 8);
    }

    // 2) h(16x256,f16) @ W_hh^T -> this wave's 8 N-tiles (128 gate cols)
    v8f acc[8] = {};
#pragma unroll
    for (int k = 0; k < 8; ++k) {
      v16h a = load_frag(h_lds, 264, 0, k * 32, lane);
#pragma unroll
      for (int n = 0; n < 8; ++n) {
        v16h b = load_frag(whh, HID, (wave * 8 + n) * 16, k * 32, lane);  // L2-resident
        acc[n] = wmma_f16(a, b, acc[n]);
      }
    }

    // 3) all DMA done (own wave), then make visible across waves
    wait_asynccnt0();
    __syncthreads();
#pragma unroll
    for (int n = 0; n < 8; ++n) {
      const int col = (wave * 8 + n) * 16 + ln;
#pragma unroll
      for (int r = 0; r < 8; ++r) {
        _Float16* g = g_lds + (r + 8 * hlf) * 1032 + col;  // each elem owned by one lane
        *g = (_Float16)((float)*g + acc[n][r]);
      }
    }
    __syncthreads();

    // 4) cell update: c = sig(f)*c + sig(i)*tanh(g); h = sig(o)*tanh(c)
#pragma unroll
    for (int j = 0; j < 16; ++j) {
      const int col = cg * 16 + j;
      const float gi = (float)g_lds[cr * 1032 + col];
      const float gf = (float)g_lds[cr * 1032 + 256 + col];
      const float gg = (float)g_lds[cr * 1032 + 512 + col];
      const float go = (float)g_lds[cr * 1032 + 768 + col];
      const float c = sigf(gf) * cst[j] + sigf(gi) * tanhf(gg);
      const float h = sigf(go) * tanhf(c);
      cst[j] = c;
      h_lds[cr * 264 + col] = (_Float16)h;
      hcat[((size_t)(b0 + cr) * SEQ + t) * 512 + dir * HID + col] = (_Float16)h;
    }
    __syncthreads();
  }
}

// ---------------- emissions: em = hcat @ lin_w^T (N padded to 16) ----------------
__global__ __launch_bounds__(256) void emissions_kernel(
    const _Float16* __restrict__ hcat, const _Float16* __restrict__ linw16,
    const float* __restrict__ lin_b, float* __restrict__ em /* [ROWS,16] */) {
  const int lane = threadIdx.x & 31, wave = threadIdx.x >> 5;
  const int m0 = (blockIdx.x * 8 + wave) * 16;
  const int hlf = lane >> 4, ln = lane & 15;
  v8f acc = {};
#pragma unroll
  for (int k = 0; k < 16; ++k) {
    v16h a = load_frag(hcat, 512, m0, k * 32, lane);
    v16h b = load_frag(linw16, 512, 0, k * 32, lane);
    acc = wmma_f16(a, b, acc);
  }
  const float bias = (ln < NTAG) ? lin_b[ln] : 0.f;
#pragma unroll
  for (int r = 0; r < 8; ++r)
    em[(size_t)(m0 + r + 8 * hlf) * 16 + ln] = acc[r] + bias;
}

// ---------------- Viterbi: one wave per batch row ----------------
__global__ __launch_bounds__(32) void viterbi_kernel(
    const float* __restrict__ em, const uint8_t* __restrict__ mask,
    const float* __restrict__ start_t, const float* __restrict__ end_t,
    const float* __restrict__ trans, uint8_t* __restrict__ hist,
    int* __restrict__ out) {
  const int b = blockIdx.x;
  const int t = threadIdx.x;
  __shared__ float sc[16];

  float tr[NTAG];
  float score = 0.f;
  if (t < NTAG) {
#pragma unroll
    for (int s = 0; s < NTAG; ++s) tr[s] = trans[s * NTAG + t];  // column t
    score = start_t[t] + em[((size_t)b * SEQ + 0) * 16 + t];
  }
  for (int s = 1; s < SEQ; ++s) {
    if (t < NTAG) sc[t] = score;
    __syncthreads();
    if (t < NTAG) {
      float best = -3.4e38f; int bi = 0;
#pragma unroll
      for (int sp = 0; sp < NTAG; ++sp) {
        const float v = sc[sp] + tr[sp];
        if (v > best) { best = v; bi = sp; }
      }
      best += em[((size_t)b * SEQ + s) * 16 + t];
      hist[((size_t)(s - 1) * BATCH + b) * NTAG + t] = (uint8_t)bi;
      if (mask[b * SEQ + s]) score = best;   // else keep old score
    }
    __syncthreads();
  }
  if (t < NTAG) sc[t] = score + end_t[t];
  __syncthreads();
  if (t == 0) {
    int tag = 0; float best = sc[0];
#pragma unroll
    for (int i = 1; i < NTAG; ++i) if (sc[i] > best) { best = sc[i]; tag = i; }
    for (int s = SEQ - 1; s >= 1; --s) {
      out[b * SEQ + s] = mask[b * SEQ + s] ? tag : 0;
      if (mask[b * SEQ + s]) tag = hist[((size_t)(s - 1) * BATCH + b) * NTAG + tag];
    }
    out[b * SEQ + 0] = mask[b * SEQ + 0] ? tag : 0;
  }
}

// ---------------- host launcher ----------------
extern "C" void kernel_launch(void* const* d_in, const int* in_sizes, int n_in,
                              void* d_out, int out_size, void* d_ws, size_t ws_size,
                              hipStream_t stream) {
  const int*   input  = (const int*)  d_in[0];
  const float* emb    = (const float*)d_in[1];
  const float* w_ih_f = (const float*)d_in[2];
  const float* w_hh_f = (const float*)d_in[3];
  const float* b_ih_f = (const float*)d_in[4];
  const float* b_hh_f = (const float*)d_in[5];
  const float* w_ih_b = (const float*)d_in[6];
  const float* w_hh_b = (const float*)d_in[7];
  const float* b_ih_b = (const float*)d_in[8];
  const float* b_hh_b = (const float*)d_in[9];
  const float* lin_w  = (const float*)d_in[10];
  const float* lin_b  = (const float*)d_in[11];
  const float* start_t= (const float*)d_in[12];
  const float* end_t  = (const float*)d_in[13];
  const float* trans  = (const float*)d_in[14];
  const uint8_t* mask = (const uint8_t*)d_in[15];   // bool array
  int* out = (int*)d_out;                            // reference output dtype is int32

  // workspace layout (≈180 MB)
  char* w = (char*)d_ws;
  size_t off = 0;
  auto alloc = [&](size_t bytes) { void* p = w + off; off = (off + bytes + 255) & ~(size_t)255; return p; };
  _Float16* X       = (_Float16*)alloc((size_t)ROWS * EMB * 2);     // 8 MB
  _Float16* wihf16_f= (_Float16*)alloc((size_t)G4 * EMB * 2);
  _Float16* wihf16_b= (_Float16*)alloc((size_t)G4 * EMB * 2);
  _Float16* whhf16_f= (_Float16*)alloc((size_t)G4 * HID * 2);
  _Float16* whhf16_b= (_Float16*)alloc((size_t)G4 * HID * 2);
  _Float16* linw16  = (_Float16*)alloc((size_t)16 * 512 * 2);
  _Float16* xp_f    = (_Float16*)alloc((size_t)ROWS * G4 * 2);      // 64 MB
  _Float16* xp_b    = (_Float16*)alloc((size_t)ROWS * G4 * 2);      // 64 MB
  _Float16* hcat    = (_Float16*)alloc((size_t)ROWS * 512 * 2);     // 32 MB
  float*    em      = (float*)   alloc((size_t)ROWS * 16 * 4);      // 2 MB
  uint8_t*  hist    = (uint8_t*) alloc((size_t)(SEQ - 1) * BATCH * NTAG);
  (void)ws_size; (void)in_sizes; (void)n_in; (void)out_size;

  // 1) weight conversions to f16
  cast_f32_f16_kernel<<<(G4*EMB + 255)/256, 256, 0, stream>>>(w_ih_f, wihf16_f, G4*EMB);
  cast_f32_f16_kernel<<<(G4*EMB + 255)/256, 256, 0, stream>>>(w_ih_b, wihf16_b, G4*EMB);
  cast_f32_f16_kernel<<<(G4*HID + 255)/256, 256, 0, stream>>>(w_hh_f, whhf16_f, G4*HID);
  cast_f32_f16_kernel<<<(G4*HID + 255)/256, 256, 0, stream>>>(w_hh_b, whhf16_b, G4*HID);
  pad_linw_kernel<<<(16*512 + 255)/256, 256, 0, stream>>>(lin_w, linw16);

  // 2) embedding gather -> f16
  embed_kernel<<<(ROWS*EMB + 255)/256, 256, 0, stream>>>(input, emb, X);

  // 3) input projection GEMMs (both directions), bias folded
  input_proj_kernel<<<dim3(4096, 2), 256, 0, stream>>>(
      X, wihf16_f, wihf16_b, b_ih_f, b_hh_f, b_ih_b, b_hh_b, xp_f, xp_b);

  // 4) persistent LSTM recurrence: 2 dirs x 4 batch tiles (async xp staging)
  lstm_rec_kernel<<<8, 256, 0, stream>>>(xp_f, xp_b, whhf16_f, whhf16_b, hcat);

  // 5) emissions GEMM (N padded to 16)
  emissions_kernel<<<256, 256, 0, stream>>>(hcat, linw16, lin_b, em);

  // 6) Viterbi decode, one wave per batch row
  viterbi_kernel<<<BATCH, 32, 0, stream>>>(em, mask, start_t, end_t, trans, hist, out);
}
